// Net_8169027797321
// MI455X (gfx1250) — matrix operand order repaired and checked by reading, not compiled, not run on hardware
//
#include <hip/hip_runtime.h>
#include <hip/hip_bf16.h>

// Problem sizes (match reference)
#define BB  256
#define TT  1024
#define FF  128
#define HH  256
#define OO  64
#define NTK 5

typedef __attribute__((ext_vector_type(16))) __bf16 v16bf;
typedef __attribute__((ext_vector_type(8)))  float  v8f;
typedef __attribute__((ext_vector_type(2)))  float  v2f;

union V16 { v16bf v; uint4 q[2]; };

#define HB_STRIDE 264   // 256 + 8 pad (bf16) -> conflict-free A-frag loads
#define XS_STRIDE 136   // 128 + 8 pad
#define WC_STRIDE 136   // W_ci^T row stride (bf16)
#define SMEM_ELEMS (2*16*HB_STRIDE + 2*16*XS_STRIDE + HH*WC_STRIDE)

__device__ __forceinline__ void loadA(V16& f, const __bf16* row, int c0) {
  f.q[0] = *(const uint4*)(row + c0);
  f.q[1] = *(const uint4*)(row + c0 + 8);
}

// fast sigmoid: v_exp + v_rcp (avoids v_div_fixup expansion on the serial path)
__device__ __forceinline__ float fast_sigmoid(float z) {
  return __builtin_amdgcn_rcpf(1.f + __expf(-z));
}
// overflow-safe fast tanh: tanh(|z|) = (1-e)/(1+e), e = exp(-2|z|) in (0,1]
__device__ __forceinline__ float fast_tanh(float z) {
  const float a = __builtin_fabsf(z);
  const float e = __expf(-2.f * a);
  const float r = (1.f - e) * __builtin_amdgcn_rcpf(1.f + e);
  return __builtin_copysignf(r, z);
}

// -----------------------------------------------------------------------------
// Persistent-RNN scan kernel.
// Grid: B/16 workgroups. Block: 512 threads = 16 wave32s.
// Wave w owns hidden columns [16w, 16w+16).
// W_ig / W_og B-fragments stay register-resident for all 1029 steps (128 VGPR).
// W_ci^T is bf16 in LDS (streamed, used only 4x/step). h is bf16 double-
// buffered in LDS; x_t slab is staged one step ahead (one barrier per step).
// h A-fragments are loaded in batches of 4 k-steps (8 ds_load_b128 in flight)
// so the LDS latency is paid once per batch, not once per k-step.
// -----------------------------------------------------------------------------
__global__ __launch_bounds__(512, 1)
void lstm_scan_kernel(const float* __restrict__ x,
                      const int*   __restrict__ lens,
                      const float* __restrict__ Wci,
                      const float* __restrict__ Wig,
                      const float* __restrict__ Wog,
                      const float* __restrict__ b_ci,  const float* __restrict__ b_ig,
                      const float* __restrict__ b_og,  const float* __restrict__ bt_ci,
                      const float* __restrict__ bt_ig, const float* __restrict__ bt_og,
                      float* __restrict__ h_out)
{
  extern __shared__ __align__(16) char smem_raw[];
  __bf16* hb = (__bf16*)smem_raw;                 // [2][16][HB_STRIDE]
  __bf16* xs = hb + 2 * 16 * HB_STRIDE;           // [2][16][XS_STRIDE]
  __bf16* wc = xs + 2 * 16 * XS_STRIDE;           // [HH][WC_STRIDE]  (W_ci^T)

  const int tid   = threadIdx.x;
  const int wave  = tid >> 5;        // 0..15 -> N tile
  const int lane  = tid & 31;
  const int l16   = lane & 15;
  const int lhi   = lane >> 4;       // 0/1 (lane half)
  const int bbase = blockIdx.x * 16;
  const int n     = wave * 16 + l16; // hidden column this lane owns

  // ---- W_ci^T -> LDS (bf16): wc[n][k] = Wci[k][n] ----
  for (int i = tid; i < HH * FF; i += 512) {
    const int nn = i & (HH - 1), k = i >> 8;
    wc[nn * WC_STRIDE + k] = (__bf16)Wci[k * HH + nn];
  }

  // ---- register-resident gate-weight B-fragments (persistent) ----
  // B layout (32x16 bf16): lane = column N; VGPR v holds rows
  // K = lhi*16 + {2v, 2v+1} within each 32-deep K step.
  v16bf big[8], bog[8];
  #pragma unroll
  for (int kk = 0; kk < 8; ++kk) {
    const int kb = kk * 32 + lhi * 16;
    V16 fi, fo;
    #pragma unroll
    for (int j = 0; j < 16; ++j) {
      fi.v[j] = (__bf16)Wig[(kb + j) * HH + n];
      fo.v[j] = (__bf16)Wog[(kb + j) * HH + n];
    }
    big[kk] = fi.v; bog[kk] = fo.v;
  }

  // per-column biases, tick-step variants pre-folded
  const float bc   = b_ci[n], bi = b_ig[n], bo = b_og[n];
  const float bc_t = bc + bt_ci[n];
  const float bi_t = bi + bt_ig[n];
  const float bo_t = bo + bt_og[n];

  // per-row (C/D layout: row M = r + 8*lhi) lengths + f32 state
  int   lenr[8];
  float cst[8], hst[8];
  #pragma unroll
  for (int r = 0; r < 8; ++r) {
    lenr[r] = lens[bbase + r + 8 * lhi];
    cst[r] = 0.f; hst[r] = 0.f;
  }

  // zero h buffers (h0 = 0) and stage x slab for t = 0
  for (int i = tid; i < 2 * 16 * HB_STRIDE; i += 512) hb[i] = (__bf16)0.f;
  for (int i = tid; i < 16 * FF; i += 512) {
    const int m = i >> 7, k = i & (FF - 1);
    xs[m * XS_STRIDE + k] = (__bf16)x[((size_t)(bbase + m) * TT + 0) * FF + k];
  }
  __syncthreads();

  int cur = 0;
  for (int t = 0; t < TT + NTK; ++t) {
    const __bf16* ha = hb + cur * 16 * HB_STRIDE + l16 * HB_STRIDE;
    const __bf16* xa = xs + (t & 1) * 16 * XS_STRIDE + l16 * XS_STRIDE;
    const __bf16* wr = wc + n * WC_STRIDE;

    // ---- gate GEMMs: aig/aog = h @ W_ig / W_og (weights from registers) ----
    // Batch the A-fragment loads 4 k-steps at a time: 8 ds_load_b128 issue
    // back-to-back into distinct registers, then 8 WMMAs drain with partial
    // s_wait_dscnt counts instead of a full LDS round-trip per k-step.
    v8f aig = {}; v8f aog = {}; v8f aci = {};
    #pragma unroll
    for (int half = 0; half < 2; ++half) {
      V16 af[4];
      #pragma unroll
      for (int j = 0; j < 4; ++j)
        loadA(af[j], ha, (half * 4 + j) * 32 + lhi * 16);
      #pragma unroll
      for (int j = 0; j < 4; ++j) {
        const int kk = half * 4 + j;
        aig = __builtin_amdgcn_wmma_f32_16x16x32_bf16(false, af[j].v, false, big[kk],
                                                      (short)0, aig, false, false);
        aog = __builtin_amdgcn_wmma_f32_16x16x32_bf16(false, af[j].v, false, bog[kk],
                                                      (short)0, aog, false, false);
      }
    }
    // ---- aci = x_t @ W_ci (both operands streamed from LDS) ----
    if (t < TT) {
      #pragma unroll
      for (int kk = 0; kk < 4; ++kk) {
        V16 a, b;
        loadA(a, xa, kk * 32 + lhi * 16);
        const int kb = kk * 32 + lhi * 16;    // B-frag: 16 consecutive K at col n
        b.q[0] = *(const uint4*)(wr + kb);
        b.q[1] = *(const uint4*)(wr + kb + 8);
        aci = __builtin_amdgcn_wmma_f32_16x16x32_bf16(false, a.v, false, b.v,
                                                      (short)0, aci, false, false);
      }
    }

    // ---- stage x slab for t+1 into the other xs buffer ----
    if (t + 1 < TT) {
      __bf16* xn = xs + ((t + 1) & 1) * 16 * XS_STRIDE;
      for (int i = tid; i < 16 * FF; i += 512) {
        const int m = i >> 7, k = i & (FF - 1);
        const float* p = &x[((size_t)(bbase + m) * TT + (t + 1)) * FF + k];
        xn[m * XS_STRIDE + k] = (__bf16)(*p);
        if (t + 2 < TT) __builtin_prefetch(p + FF, 0, 1);  // pull t+2 toward L2
      }
    }

    // ---- branchless elementwise gates + state update; write h to other buffer ----
    __bf16* hn = hb + (cur ^ 1) * 16 * HB_STRIDE;
    #pragma unroll
    for (int r = 0; r < 8; ++r) {
      const int   M   = r + 8 * lhi;
      const int   len = lenr[r];
      const bool  reg = (t < len);
      const bool  act = (t < len + NTK);        // reg || tick
      const float ci = fast_tanh(reg ? (aci[r] + bc) : bc_t);
      const float ig = fast_sigmoid(aig[r] + (reg ? bi : bi_t));
      const float og = fast_sigmoid(aog[r] + (reg ? bo : bo_t));
      const float cn = cst[r] + ci * ig;        // no forget gate: pure accumulation
      const float hv = cn * og;
      cst[r] = act ? cn : cst[r];               // v_cndmask, no branch
      hst[r] = act ? hv : hst[r];
      hn[M * HB_STRIDE + n] = (__bf16)hst[r];
    }
    __syncthreads();                            // single barrier per step
    cur ^= 1;
  }

  // final hidden state (f32) for the FC epilogue
  #pragma unroll
  for (int r = 0; r < 8; ++r)
    h_out[(size_t)(bbase + r + 8 * lhi) * HH + n] = hst[r];
}

// -----------------------------------------------------------------------------
// FC epilogue: out = h_final @ W_fc + b_fc, exact f32 via V_WMMA_F32_16X16X4_F32.
// Grid (B/16, O/16), one wave32 per 16x16 tile.
// -----------------------------------------------------------------------------
__global__ __launch_bounds__(32)
void fc_kernel(const float* __restrict__ h,    // [B][H]
               const float* __restrict__ Wfc,  // [H][O]
               const float* __restrict__ bfc,  // [O]
               float* __restrict__ out)        // [B][O]
{
  const int lane = threadIdx.x & 31;
  const int l16  = lane & 15, lhi = lane >> 4;
  const int mb   = blockIdx.x * 16;
  const int nb   = blockIdx.y * 16;

  v8f acc = {};
  #pragma unroll 8
  for (int kb = 0; kb < HH; kb += 4) {
    const int k0 = kb + 2 * lhi;                 // f32 A/B layout: lane half -> K pair
    v2f a = *(const v2f*)&h[(size_t)(mb + l16) * HH + k0];
    v2f b;
    b[0] = Wfc[(k0    ) * OO + nb + l16];
    b[1] = Wfc[(k0 + 1) * OO + nb + l16];
    acc = __builtin_amdgcn_wmma_f32_16x16x4_f32(false, a, false, b,
                                                (short)0, acc, false, false);
  }
  const float bias = bfc[nb + l16];
  #pragma unroll
  for (int r = 0; r < 8; ++r)
    out[(size_t)(mb + r + 8 * lhi) * OO + nb + l16] = acc[r] + bias;
}

// -----------------------------------------------------------------------------
extern "C" void kernel_launch(void* const* d_in, const int* in_sizes, int n_in,
                              void* d_out, int out_size, void* d_ws, size_t ws_size,
                              hipStream_t stream) {
  const float* x     = (const float*)d_in[0];
  const int*   lens  = (const int*)  d_in[1];
  const float* Wci   = (const float*)d_in[2];
  const float* Wig   = (const float*)d_in[3];
  const float* Wog   = (const float*)d_in[4];
  const float* b_ci  = (const float*)d_in[5];
  const float* b_ig  = (const float*)d_in[6];
  const float* b_og  = (const float*)d_in[7];
  const float* bt_ci = (const float*)d_in[8];
  const float* bt_ig = (const float*)d_in[9];
  const float* bt_og = (const float*)d_in[10];
  const float* Wfc   = (const float*)d_in[11];
  const float* bfc   = (const float*)d_in[12];

  float* h_out = (float*)d_ws;          // B*H f32 = 256 KB scratch
  float* out   = (float*)d_out;         // [B][O] f32

  const size_t smem = SMEM_ELEMS * sizeof(__bf16);
  lstm_scan_kernel<<<BB / 16, 512, smem, stream>>>(
      x, lens, Wci, Wig, Wog, b_ci, b_ig, b_og, bt_ci, bt_ig, bt_og, h_out);

  dim3 g(BB / 16, OO / 16);
  fc_kernel<<<g, 32, 0, stream>>>(h_out, Wfc, bfc, out);
}
